// BranchedCIMNet_79190607003770
// MI455X (gfx1250) — matrix operand order repaired and checked by reading, not compiled, not run on hardware
//
#include <hip/hip_runtime.h>

// ---------------------------------------------------------------------------
// Types for CDNA5 WMMA (wave32, V_WMMA_F32_16X16X32_F16)
// ---------------------------------------------------------------------------
typedef _Float16 h16;
typedef __attribute__((ext_vector_type(8)))  _Float16 v8h;
typedef __attribute__((ext_vector_type(16))) _Float16 v16h;
typedef __attribute__((ext_vector_type(8)))  float    v8f;
typedef int v4i __attribute__((vector_size(16)));   // param type of async-LDS builtins

#define BM 128
#define BN 128
#define BK 32
#define LDSS 40   // LDS row stride in halves (32 data + 8 pad -> conflict-free, 16B aligned)

// gfx1250 async global->LDS copy path (ASYNCcnt-tracked), with fallback.
#if __has_builtin(__builtin_amdgcn_global_load_async_to_lds_b128) && \
    __has_builtin(__builtin_amdgcn_s_wait_asynccnt)
#define USE_ASYNC_LDS 1
#else
#define USE_ASYNC_LDS 0
#endif

// ---------------------------------------------------------------------------
// Layout transform: NCHW fp32 -> NHWC f16
// ---------------------------------------------------------------------------
__global__ __launch_bounds__(256)
void nchw_to_nhwc_f16(const float* __restrict__ in, h16* __restrict__ out,
                      int N, int C, int H, int W)
{
  size_t total = (size_t)N * C * H * W;
  size_t i = (size_t)blockIdx.x * 256 + threadIdx.x;
  if (i >= total) return;
  int c = (int)(i % C); size_t r = i / C;
  int w = (int)(r % W); r /= W;
  int h = (int)(r % H); int n = (int)(r / H);
  out[i] = (h16)in[(((size_t)n * C + c) * H + h) * W + w];
}

// ---------------------------------------------------------------------------
// Fold BN into conv weights: OIHW fp32 -> [Cout][KH*KW*Cin] f16, bias fp32
// ---------------------------------------------------------------------------
__global__ __launch_bounds__(256)
void convert_conv_weight(const float* __restrict__ w,
                         const float* __restrict__ gam, const float* __restrict__ bet,
                         const float* __restrict__ mu,  const float* __restrict__ var,
                         h16* __restrict__ wout, float* __restrict__ bout,
                         int Cout, int Cin, int KH, int KW)
{
  int K = KH * KW * Cin;
  size_t total = (size_t)Cout * K;
  size_t i = (size_t)blockIdx.x * 256 + threadIdx.x;
  if (i < total) {
    int co = (int)(i / K); int r = (int)(i % K);
    int p = r / Cin; int ci = r % Cin;
    int kh = p / KW; int kw = p % KW;
    float inv = gam[co] * rsqrtf(var[co] + 1e-5f);
    wout[i] = (h16)(w[(((size_t)co * Cin + ci) * KH + kh) * KW + kw] * inv);
  }
  if (i < (size_t)Cout) {
    float inv = gam[i] * rsqrtf(var[i] + 1e-5f);
    bout[i] = bet[i] - mu[i] * inv;
  }
}

// ---------------------------------------------------------------------------
// Implicit-GEMM convolution on WMMA.
//   M = Nb*Hout*Wout rows (NHWC), N = Cout, K = KH*KW*Cin.
//   128x128 block tile, 8 waves, each wave = 64x32 (4x2 WMMA 16x16 accs).
//   Tiles staged through LDS via async global->LDS copies (ASYNCcnt) when
//   available; edge/zero-fill cases use ds stores (DScnt). Both counters are
//   drained before the workgroup barrier.
//   Fused: +bias (BN folded), +residual, ReLU. f16 in/out, f32 accumulate.
// ---------------------------------------------------------------------------
__global__ __launch_bounds__(256)
void conv_igemm_wmma(const h16* __restrict__ in, const h16* __restrict__ wgt,
                     const float* __restrict__ bias, const h16* __restrict__ residual,
                     h16* __restrict__ out,
                     int Nb, int H, int W, int Cin, int Hout, int Wout, int Cout,
                     int KH, int KW, int stride, int pad, int relu)
{
  __shared__ h16 lds_a[BM * LDSS];
  __shared__ h16 lds_b[BN * LDSS];

  const int tid  = threadIdx.x;
  const int lane = tid & 31;
  const int wave = tid >> 5;
  const int M    = Nb * Hout * Wout;
  const int Kg   = KH * KW * Cin;
  const int m0   = blockIdx.x * BM;
  const int n0   = blockIdx.y * BN;
  const int wm   = (wave >> 2) * 64;   // 2 waves along M
  const int wn   = (wave & 3) * 32;    // 4 waves along N
  const int lrow = lane & 15;
  const int hi16 = lane >> 4;          // 0 or 1

  v8f acc[4][2] = {};

  // cooperative tile-load mapping: 256 threads, each owns 16 halves of a row
  const int row  = tid >> 1;           // 0..127
  const int kseg = (tid & 1) * 16;     // 0 or 16
  const bool fastA = (Cin % 8) == 0;   // every conv except the 7x7 stem
  const bool fastB = (Kg  % 8) == 0;

  int ow = 0, oh = 0, nb = 0;
  const int mrow = m0 + row;
  const bool mok = (mrow < M);
  if (mok) { ow = mrow % Wout; int t = mrow / Wout; oh = t % Hout; nb = t / Hout; }
  const int  nrow = n0 + row;
  const bool nok  = (nrow < Cout);

  for (int k0 = 0; k0 < Kg; k0 += BK) {
    // ---- stage A tile (im2col gather of activations) ----
    {
      h16* dst = &lds_a[row * LDSS + kseg];
      #pragma unroll
      for (int hseg = 0; hseg < 2; ++hseg) {
        int kk = k0 + kseg + hseg * 8;
        if (mok && fastA && (kk + 7 < Kg)) {
          int p  = kk / Cin, ci = kk % Cin;
          int kh = p / KW,   kw = p % KW;
          int ih = oh * stride - pad + kh;
          int iw = ow * stride - pad + kw;
          if ((unsigned)ih < (unsigned)H && (unsigned)iw < (unsigned)W) {
            const h16* src = in + ((((size_t)nb * H + ih) * W + iw) * Cin + ci);
#if USE_ASYNC_LDS
            __builtin_amdgcn_global_load_async_to_lds_b128(
                (v4i*)src, (v4i*)(dst + hseg * 8), 0, 0);
#else
            *(v8h*)(dst + hseg * 8) = *(const v8h*)src;
#endif
          } else {
            v8h z = {}; *(v8h*)(dst + hseg * 8) = z;
          }
        } else {
          #pragma unroll
          for (int j = 0; j < 8; ++j) {
            int k = kk + j;
            h16 v = (h16)0.f;
            if (mok && k < Kg) {
              int p  = k / Cin, ci = k % Cin;
              int kh = p / KW,  kw = p % KW;
              int ih = oh * stride - pad + kh;
              int iw = ow * stride - pad + kw;
              if ((unsigned)ih < (unsigned)H && (unsigned)iw < (unsigned)W)
                v = in[(((size_t)nb * H + ih) * W + iw) * Cin + ci];
            }
            dst[hseg * 8 + j] = v;
          }
        }
      }
    }
    // ---- stage B tile (weights [Cout][Kg], K-contiguous) ----
    {
      h16* dst = &lds_b[row * LDSS + kseg];
      if (nok && (k0 + BK < Kg))
        __builtin_prefetch(wgt + (size_t)nrow * Kg + k0 + BK, 0, 0);
      #pragma unroll
      for (int hseg = 0; hseg < 2; ++hseg) {
        int kk = k0 + kseg + hseg * 8;
        if (nok && fastB && (kk + 7 < Kg)) {
          const h16* src = wgt + (size_t)nrow * Kg + kk;
#if USE_ASYNC_LDS
          __builtin_amdgcn_global_load_async_to_lds_b128(
              (v4i*)src, (v4i*)(dst + hseg * 8), 0, 0);
#else
          *(v8h*)(dst + hseg * 8) = *(const v8h*)src;
#endif
        } else {
          #pragma unroll
          for (int j = 0; j < 8; ++j) {
            int k = kk + j;
            dst[hseg * 8 + j] = (nok && k < Kg) ? wgt[(size_t)nrow * Kg + k] : (h16)0.f;
          }
        }
      }
    }
#if USE_ASYNC_LDS
    __builtin_amdgcn_s_wait_asynccnt(0);   // drain async global->LDS copies
#endif
    __syncthreads();                        // also drains DScnt for edge fills

    // ---- 8 WMMAs per wave per K-step ----
    const int kbA = hi16 * 8;  // ISA layout: lanes 0-15 hold K 0-7/16-23, 16-31 hold K 8-15/24-31
    #pragma unroll
    for (int mi = 0; mi < 4; ++mi) {
      const h16* ra = &lds_a[(wm + mi * 16 + lrow) * LDSS];
      v8h alo = *(const v8h*)(ra + kbA);
      v8h ahi = *(const v8h*)(ra + kbA + 16);
      v16h af;
      #pragma unroll
      for (int j = 0; j < 8; ++j) { af[j] = alo[j]; af[j + 8] = ahi[j]; }
      #pragma unroll
      for (int ni = 0; ni < 2; ++ni) {
        const h16* rb = &lds_b[(wn + ni * 16 + lrow) * LDSS + hi16 * 8];
        v8h blo = *(const v8h*)(rb);
        v8h bhi = *(const v8h*)(rb + 16);
        v16h bf;
        #pragma unroll
        for (int j = 0; j < 8; ++j) { bf[j] = blo[j]; bf[j + 8] = bhi[j]; }
        acc[mi][ni] = __builtin_amdgcn_wmma_f32_16x16x32_f16(
            false, af, false, bf, (short)0, acc[mi][ni], false, false);
      }
    }
    __syncthreads();
  }

  // ---- epilogue: bias + residual + ReLU, f32 -> f16 NHWC store ----
  #pragma unroll
  for (int mi = 0; mi < 4; ++mi) {
    #pragma unroll
    for (int ni = 0; ni < 2; ++ni) {
      #pragma unroll
      for (int r = 0; r < 8; ++r) {
        int m = m0 + wm + mi * 16 + r + hi16 * 8;  // C/D layout: lanes16-31 -> M=r+8
        int n = n0 + wn + ni * 16 + lrow;
        if (m < M && n < Cout) {
          float v = acc[mi][ni][r] + bias[n];
          if (residual) v += (float)residual[(size_t)m * Cout + n];
          if (relu) v = fmaxf(v, 0.f);
          out[(size_t)m * Cout + n] = (h16)v;
        }
      }
    }
  }
}

// ---------------------------------------------------------------------------
// 3x3 stride-2 pad-1 maxpool, NHWC f16
// ---------------------------------------------------------------------------
__global__ __launch_bounds__(256)
void maxpool3x3s2_nhwc(const h16* __restrict__ in, h16* __restrict__ out,
                       int N, int C, int H, int W, int Hout, int Wout)
{
  size_t total = (size_t)N * Hout * Wout * C;
  size_t i = (size_t)blockIdx.x * 256 + threadIdx.x;
  if (i >= total) return;
  int c = (int)(i % C); size_t r = i / C;
  int ow = (int)(r % Wout); r /= Wout;
  int oh = (int)(r % Hout); int n = (int)(r / Hout);
  float m = -3.4e38f;
  #pragma unroll
  for (int dh = 0; dh < 3; ++dh) {
    int ih = oh * 2 - 1 + dh;
    if ((unsigned)ih >= (unsigned)H) continue;
    #pragma unroll
    for (int dw = 0; dw < 3; ++dw) {
      int iw = ow * 2 - 1 + dw;
      if ((unsigned)iw >= (unsigned)W) continue;
      m = fmaxf(m, (float)in[(((size_t)n * H + ih) * W + iw) * C + c]);
    }
  }
  out[i] = (h16)m;
}

// ---------------------------------------------------------------------------
// Global average pool: [N, HW, C] f16 -> [N, C] fp32
// ---------------------------------------------------------------------------
__global__ __launch_bounds__(256)
void avgpool_nhwc(const h16* __restrict__ in, float* __restrict__ out,
                  int N, int HW, int C)
{
  int i = blockIdx.x * 256 + threadIdx.x;
  if (i >= N * C) return;
  int n = i / C, c = i % C;
  float s = 0.f;
  for (int t = 0; t < HW; ++t) s += (float)in[((size_t)n * HW + t) * C + c];
  out[i] = s / (float)HW;
}

// ---------------------------------------------------------------------------
// Head: fc(512->512) + cmd MLP(5->256->256) + hard-routed expert MLP.
// One block per batch row; tiny FLOPs -> plain fp32 VALU.
// ---------------------------------------------------------------------------
__global__ __launch_bounds__(256)
void head_kernel(const float* __restrict__ pooled, const float* __restrict__ x2,
                 const float* __restrict__ fcW, const float* __restrict__ fcb,
                 const float* __restrict__ cW1, const float* __restrict__ cb1,
                 const float* __restrict__ cW2, const float* __restrict__ cb2,
                 const float* __restrict__ eW1, const float* __restrict__ eb1,
                 const float* __restrict__ eW2, const float* __restrict__ eb2,
                 const float* __restrict__ eW3, const float* __restrict__ eb3,
                 float* __restrict__ out)
{
  __shared__ float f[768];
  __shared__ float h1[256];
  __shared__ float h2[256];
  __shared__ int sidx;
  const int b = blockIdx.x;
  const int t = threadIdx.x;

  // fc: feat_rgb (no activation, matches reference)
  for (int o = t; o < 512; o += 256) {
    float s = fcb[o];
    for (int i = 0; i < 512; ++i) s += pooled[b * 512 + i] * fcW[(size_t)i * 512 + o];
    f[o] = s;
  }
  // cmd layer 1
  if (t < 256) {
    float s = cb1[t];
    for (int i = 0; i < 5; ++i) s += x2[b * 5 + i] * cW1[i * 256 + t];
    h1[t] = fmaxf(s, 0.f);
  }
  __syncthreads();
  // cmd layer 2 -> f[512..767]
  if (t < 256) {
    float s = cb2[t];
    for (int i = 0; i < 256; ++i) s += h1[i] * cW2[(size_t)i * 256 + t];
    f[512 + t] = fmaxf(s, 0.f);
  }
  if (t == 0) {
    int best = 0; float bv = x2[b * 5];
    for (int i = 1; i < 5; ++i) { float v = x2[b * 5 + i]; if (v > bv) { bv = v; best = i; } }
    sidx = best;
  }
  __syncthreads();
  const int idx = sidx;
  if (t < 256) {
    float s = eb1[idx * 256 + t];
    for (int i = 0; i < 768; ++i) s += f[i] * eW1[((size_t)idx * 768 + i) * 256 + t];
    h2[t] = fmaxf(s, 0.f);
  }
  __syncthreads();
  if (t < 256) {
    float s = eb2[idx * 256 + t];
    for (int i = 0; i < 256; ++i) s += h2[i] * eW2[((size_t)idx * 256 + i) * 256 + t];
    h1[t] = fmaxf(s, 0.f);
  }
  __syncthreads();
  if (t < 3) {
    float s = eb3[idx * 3 + t];
    for (int i = 0; i < 256; ++i) s += h1[i] * eW3[((size_t)idx * 256 + i) * 3 + t];
    out[b * 3 + t] = s;
  }
}

// ---------------------------------------------------------------------------
// Host orchestration
// ---------------------------------------------------------------------------
struct BnP  { const float* p[4]; };
struct BlkP { BnP bn1, bn2; const float* c1; const float* c2; BnP dbn; const float* dw; int down; };

static inline void launch_conv(hipStream_t stream, const h16* in, const h16* w,
                               const float* bias, const h16* resid, h16* out,
                               int Nb, int H, int W, int Cin, int Hout, int Wout,
                               int Cout, int KH, int KW, int stride, int pad, int relu)
{
  int M = Nb * Hout * Wout;
  dim3 grid((M + BM - 1) / BM, (Cout + BN - 1) / BN);
  conv_igemm_wmma<<<grid, 256, 0, stream>>>(in, w, bias, resid, out,
      Nb, H, W, Cin, Hout, Wout, Cout, KH, KW, stride, pad, relu);
}

static inline void launch_wconv(hipStream_t stream, const float* w, const BnP& bn,
                                h16* wout, float* bout, int Cout, int Cin, int KH, int KW)
{
  size_t total = (size_t)Cout * KH * KW * Cin;
  int blocks = (int)((total + 255) / 256);
  convert_conv_weight<<<blocks, 256, 0, stream>>>(w, bn.p[0], bn.p[1], bn.p[2], bn.p[3],
                                                  wout, bout, Cout, Cin, KH, KW);
}

extern "C" void kernel_launch(void* const* d_in, const int* in_sizes, int n_in,
                              void* d_out, int out_size, void* d_ws, size_t ws_size,
                              hipStream_t stream)
{
  (void)in_sizes; (void)out_size; (void)ws_size;
  if (n_in < 194) return;  // expected leaf count for this pytree

  const int NB = 64;
  static const int BLOCKS[4] = {3, 4, 6, 3};
  static const int CHN[4]    = {64, 128, 256, 512};

  // ---- walk inputs (JAX pytree: dicts sorted by key, tuples/lists in order) ----
  const float* x1 = (const float*)d_in[0];
  const float* x2 = (const float*)d_in[1];
  int cur = 2;
  auto nxt = [&]() { return (const float*)d_in[cur++]; };

  BnP bn1; for (int i = 0; i < 4; ++i) bn1.p[i] = nxt();                 // params.bn1
  const float *cW1 = nxt(), *cb1 = nxt(), *cW2 = nxt(), *cb2 = nxt();    // params.cmd
  const float* conv1_w = nxt();                                          // params.conv1
  const float *eW1 = nxt(), *eW2 = nxt(), *eW3 = nxt();                  // params.experts (W1,W2,W3,...)
  const float *eb1 = nxt(), *eb2 = nxt(), *eb3 = nxt();
  const float *fcW = nxt(), *fcb = nxt();                                // params.fc
  BlkP blk[16]; int nblk = 0;
  for (int s = 0; s < 4; ++s) {
    for (int b = 0; b < BLOCKS[s]; ++b) {
      BlkP& B = blk[nblk++];
      for (int i = 0; i < 4; ++i) B.bn1.p[i] = nxt();
      for (int i = 0; i < 4; ++i) B.bn2.p[i] = nxt();
      B.c1 = nxt(); B.c2 = nxt();
      B.down = (s > 0 && b == 0);
      B.dw = nullptr;
      if (B.down) { for (int i = 0; i < 4; ++i) B.dbn.p[i] = nxt(); B.dw = nxt(); }
    }
  }

  // ---- workspace carve ----
  uint8_t* ws = (uint8_t*)d_ws;
  size_t off = 0;
  auto walloc = [&](size_t bytes) -> void* {
    off = (off + 255) & ~(size_t)255;
    void* p = ws + off; off += bytes; return p;
  };

  h16*   inbuf  = (h16*)walloc((size_t)NB * 224 * 224 * 3 * sizeof(h16));
  h16*   bigbuf = (h16*)walloc((size_t)NB * 112 * 112 * 64 * sizeof(h16));
  size_t SBUF   = (size_t)NB * 56 * 56 * 64 * sizeof(h16);
  h16*   bufs[4];
  for (int i = 0; i < 4; ++i) bufs[i] = (h16*)walloc(SBUF);
  float* pooled = (float*)walloc((size_t)NB * 512 * sizeof(float));

  // ---- stem: NCHW->NHWC f16, conv1 7x7/2 + BN + ReLU, maxpool 3x3/2 ----
  {
    size_t tot = (size_t)NB * 3 * 224 * 224;
    nchw_to_nhwc_f16<<<(unsigned)((tot + 255) / 256), 256, 0, stream>>>(x1, inbuf, NB, 3, 224, 224);
  }
  {
    h16*   w0 = (h16*)walloc((size_t)64 * 3 * 7 * 7 * sizeof(h16));
    float* b0 = (float*)walloc(64 * sizeof(float));
    launch_wconv(stream, conv1_w, bn1, w0, b0, 64, 3, 7, 7);
    launch_conv(stream, inbuf, w0, b0, nullptr, bigbuf,
                NB, 224, 224, 3, 112, 112, 64, 7, 7, 2, 3, 1);
  }
  {
    size_t tot = (size_t)NB * 56 * 56 * 64;
    maxpool3x3s2_nhwc<<<(unsigned)((tot + 255) / 256), 256, 0, stream>>>(
        bigbuf, bufs[0], NB, 64, 112, 112, 56, 56);
  }

  // ---- residual stages ----
  int curIdx = 0, curH = 56, curC = 64, bi = 0;
  for (int s = 0; s < 4; ++s) {
    for (int b = 0; b < BLOCKS[s]; ++b, ++bi) {
      const BlkP& B = blk[bi];
      const int Cin = curC, Cout = CHN[s];
      const int stride = B.down ? 2 : 1;
      const int Hin = curH, Hout = Hin / stride;

      int fr[3], fi = 0;
      for (int i = 0; i < 4; ++i) if (i != curIdx) fr[fi++] = i;
      h16* t1 = bufs[fr[0]];
      h16* db = bufs[fr[1]];
      h16* ot = bufs[fr[2]];
      const h16* curp = bufs[curIdx];

      h16*   w1 = (h16*)walloc((size_t)Cout * Cin * 9 * sizeof(h16));
      float* a1 = (float*)walloc(Cout * sizeof(float));
      launch_wconv(stream, B.c1, B.bn1, w1, a1, Cout, Cin, 3, 3);
      launch_conv(stream, curp, w1, a1, nullptr, t1,
                  NB, Hin, Hin, Cin, Hout, Hout, Cout, 3, 3, stride, 1, 1);

      const h16* resid = curp;
      if (B.down) {
        h16*   wd = (h16*)walloc((size_t)Cout * Cin * sizeof(h16));
        float* ad = (float*)walloc(Cout * sizeof(float));
        launch_wconv(stream, B.dw, B.dbn, wd, ad, Cout, Cin, 1, 1);
        launch_conv(stream, curp, wd, ad, nullptr, db,
                    NB, Hin, Hin, Cin, Hout, Hout, Cout, 1, 1, stride, 0, 0);
        resid = db;
      }

      h16*   w2 = (h16*)walloc((size_t)Cout * Cout * 9 * sizeof(h16));
      float* a2 = (float*)walloc(Cout * sizeof(float));
      launch_wconv(stream, B.c2, B.bn2, w2, a2, Cout, Cout, 3, 3);
      launch_conv(stream, t1, w2, a2, resid, ot,
                  NB, Hout, Hout, Cout, Hout, Hout, Cout, 3, 3, 1, 1, 1);

      curIdx = fr[2]; curH = Hout; curC = Cout;
    }
  }

  // ---- head ----
  {
    int tot = NB * 512;
    avgpool_nhwc<<<(tot + 255) / 256, 256, 0, stream>>>(bufs[curIdx], pooled, NB, 7 * 7, 512);
  }
  head_kernel<<<NB, 256, 0, stream>>>(pooled, x2, fcW, fcb,
                                      cW1, cb1, cW2, cb2,
                                      eW1, eb1, eW2, eb2, eW3, eb3,
                                      (float*)d_out);
}
